// GraphSAGE_14783277433239
// MI455X (gfx1250) — matrix operand order repaired and checked by reading, not compiled, not run on hardware
//
#include <hip/hip_runtime.h>
#include <hip/hip_bf16.h>

// GraphSAGE 2-layer for MI455X (gfx1250, wave32, WMMA).
//
// Strategy (memory-roofline driven):
//   - Project features BEFORE edge aggregation (linearity of mean+matmul):
//     scatters move HIDDEN=64 (layer0) / OUT=32 (layer1) floats per edge
//     instead of 128/64 -> dominant HBM traffic halved.
//   - Node GEMMs use v_wmma_f32_16x16x32_bf16 (f32 accum).
//   - W is pre-packed into LDS in *fragment-major* per-lane layout so each
//     B fragment is one aligned 32B vector load (2x ds_load_b128), not 16
//     scalar ds_load_u16 + v_mov_b16 merges.
//   - LayerNorm+ReLU fused, one wave32 per node, shfl_xor reduction.

typedef __attribute__((ext_vector_type(16))) __bf16 v16bf;
typedef __attribute__((ext_vector_type(8)))  float  v8f;

#define N_NODES 100000
#define N_EDGES 1600000
#define IN_DIM  128
#define HID_DIM 64
#define OUT_DIM 32

// ---------------------------------------------------------------- fp32->bf16
__global__ __launch_bounds__(256) void cvt_bf16_kernel(
    const float* __restrict__ in, __bf16* __restrict__ out, long long n) {
  long long i = ((long long)blockIdx.x * 256 + threadIdx.x) * 4;
  if (i + 3 < n) {
    float4 v = *(const float4*)(in + i);
    out[i + 0] = (__bf16)v.x;
    out[i + 1] = (__bf16)v.y;
    out[i + 2] = (__bf16)v.z;
    out[i + 3] = (__bf16)v.w;
  } else {
    for (; i < n; ++i) out[i] = (__bf16)in[i];
  }
}

// ---------------------------------------------------------------- degrees
__global__ __launch_bounds__(256) void degree_kernel(
    const long long* __restrict__ ei, float* __restrict__ cnt, int nEdges) {
  int e = blockIdx.x * 256 + threadIdx.x;
  if (e < nEdges) {
    long long d = ei[(long long)nEdges + e];  // dst row of edge_index
    atomicAdd(&cnt[d], 1.0f);
  }
}

// ---------------------------------------------------------------- WMMA GEMM
// S = A @ Wself,  P = A @ Wneigh.   A: [nNodes, KDIM] bf16 row-major.
// W*: [KDIM, NDIM] f32 row-major, permuted to per-lane fragment layout in LDS.
// One wave32 computes a 16-node x NDIM tile; A fragments reused across
// n-tiles and across the two B matrices.
template <int KDIM, int NDIM>
__global__ __launch_bounds__(256) void sage_gemm_kernel(
    const __bf16* __restrict__ A, const float* __restrict__ Wself,
    const float* __restrict__ Wneigh, float* __restrict__ S,
    float* __restrict__ P, int nNodes) {
  constexpr int KT = KDIM / 32;  // k-tiles of 32
  constexpr int NT = NDIM / 16;  // n-tiles of 16

  // Fragment-major B storage: [kt][nt][lane][e], 32B (one v16bf) per lane.
  __shared__ v16bf lWs[KT * NT * 32];
  __shared__ v16bf lWn[KT * NT * 32];
  {
    __bf16* ps = (__bf16*)lWs;
    __bf16* pn = (__bf16*)lWn;
    for (int i = threadIdx.x; i < KDIM * NDIM; i += 256) {
      const int e   = i & 15;          // element within fragment
      const int ln  = (i >> 4) & 31;   // lane
      const int til = i >> 9;          // fragment tile
      const int nt  = til % NT;
      const int kt  = til / NT;
      // ISA B layout (16-bit, 32x16): lane half splits K by 8s (matches A).
      const int k = kt * 32 + (e < 8 ? 8 * (ln >> 4) + e : 16 + 8 * (ln >> 4) + (e - 8));
      const int n = nt * 16 + (ln & 15);
      ps[i] = (__bf16)Wself[k * NDIM + n];
      pn[i] = (__bf16)Wneigh[k * NDIM + n];
    }
  }
  __syncthreads();

  const int lane = threadIdx.x & 31;
  const int wave = threadIdx.x >> 5;
  const int tile = blockIdx.x * 8 + wave;   // 16 nodes per tile
  if (tile * 16 >= nNodes) return;          // wave-uniform exit
  const int node0 = tile * 16;
  const int half  = lane >> 4;              // lane half selects K sub-range
  const int m     = lane & 15;              // A: row; B/D: column

  // --- A fragments (ISA 16-bit A 16x32 layout): lane half splits K by 8s
  v16bf a[KT];
  const __bf16* arow = A + (size_t)(node0 + m) * KDIM;
#pragma unroll
  for (int kt = 0; kt < KT; ++kt) {
    const int kb = kt * 32;
#pragma unroll
    for (int i = 0; i < 8; ++i) {
      a[kt][i]     = arow[kb + 8 * half + i];        // K = i   + 8*half
      a[kt][i + 8] = arow[kb + 16 + 8 * half + i];   // K = i+16+ 8*half
    }
  }

#pragma unroll
  for (int nt = 0; nt < NT; ++nt) {
    const int n0 = nt * 16;
    v8f accS = {};
    v8f accN = {};
#pragma unroll
    for (int kt = 0; kt < KT; ++kt) {
      const int fo = (kt * NT + nt) * 32 + lane;  // fragment offset
      v16bf bs = lWs[fo];   // one 32B vector load -> 2x ds_load_b128
      v16bf bn = lWn[fo];
      accS = __builtin_amdgcn_wmma_f32_16x16x32_bf16(
          false, a[kt], false, bs, (short)0, accS, false, false);
      accN = __builtin_amdgcn_wmma_f32_16x16x32_bf16(
          false, a[kt], false, bn, (short)0, accN, false, false);
    }
    // D layout: vgpr r -> row (r + 8*half), lane -> column
#pragma unroll
    for (int r = 0; r < 8; ++r) {
      const size_t o = (size_t)(node0 + r + 8 * half) * NDIM + n0 + m;
      S[o] = accS[r];
      P[o] = accN[r];
    }
  }
}

// ---------------------------------------------------------------- scatter-add
// agg[dst] += feat[src], F floats per edge, F/4 lanes per edge, float4 reads.
template <int F>
__global__ __launch_bounds__(256) void scatter_add_kernel(
    const float* __restrict__ feat, const long long* __restrict__ ei,
    float* __restrict__ agg, int nEdges) {
  constexpr int TPE = F / 4;
  long long gid = (long long)blockIdx.x * 256 + threadIdx.x;
  long long e = gid / TPE;
  int c = (int)(gid % TPE);
  if (e >= nEdges) return;
  long long s = ei[e];
  long long d = ei[(long long)nEdges + e];
  float4 v = *(const float4*)(feat + (size_t)s * F + c * 4);
  float* dp = agg + (size_t)d * F + c * 4;
  atomicAdd(dp + 0, v.x);
  atomicAdd(dp + 1, v.y);
  atomicAdd(dp + 2, v.z);
  atomicAdd(dp + 3, v.w);
}

// ---------------------------------------------------------------- LN+ReLU, 64
// z = S + agg/max(cnt,1) + b ; layernorm over 64; *g + be; relu; -> bf16 h
__global__ __launch_bounds__(256) void combine_ln64_kernel(
    const float* __restrict__ S, const float* __restrict__ agg,
    const float* __restrict__ cnt, const float* __restrict__ b,
    const float* __restrict__ g, const float* __restrict__ be,
    __bf16* __restrict__ h, int nNodes) {
  int node = blockIdx.x * 8 + (threadIdx.x >> 5);
  if (node >= nNodes) return;
  int lane = threadIdx.x & 31;
  float inv = 1.0f / fmaxf(cnt[node], 1.0f);
  size_t base = (size_t)node * 64;
  float z0 = S[base + lane]      + agg[base + lane]      * inv + b[lane];
  float z1 = S[base + lane + 32] + agg[base + lane + 32] * inv + b[lane + 32];
  float s = z0 + z1, ss = z0 * z0 + z1 * z1;
  for (int off = 16; off > 0; off >>= 1) {
    s  += __shfl_xor(s, off, 32);
    ss += __shfl_xor(ss, off, 32);
  }
  float mu  = s * (1.0f / 64.0f);
  float var = ss * (1.0f / 64.0f) - mu * mu;
  float r = rsqrtf(var + 1e-5f);
  float y0 = fmaxf((z0 - mu) * r * g[lane]      + be[lane],      0.0f);
  float y1 = fmaxf((z1 - mu) * r * g[lane + 32] + be[lane + 32], 0.0f);
  h[base + lane]      = (__bf16)y0;
  h[base + lane + 32] = (__bf16)y1;
}

// ---------------------------------------------------------------- LN+ReLU, 32
__global__ __launch_bounds__(256) void combine_ln32_kernel(
    const float* __restrict__ S, const float* __restrict__ agg,
    const float* __restrict__ cnt, const float* __restrict__ b,
    const float* __restrict__ g, const float* __restrict__ be,
    float* __restrict__ out, int nNodes) {
  int node = blockIdx.x * 8 + (threadIdx.x >> 5);
  if (node >= nNodes) return;
  int lane = threadIdx.x & 31;
  float inv = 1.0f / fmaxf(cnt[node], 1.0f);
  size_t base = (size_t)node * 32;
  float z = S[base + lane] + agg[base + lane] * inv + b[lane];
  float s = z, ss = z * z;
  for (int off = 16; off > 0; off >>= 1) {
    s  += __shfl_xor(s, off, 32);
    ss += __shfl_xor(ss, off, 32);
  }
  float mu  = s * (1.0f / 32.0f);
  float var = ss * (1.0f / 32.0f) - mu * mu;
  float r = rsqrtf(var + 1e-5f);
  out[base + lane] = fmaxf((z - mu) * r * g[lane] + be[lane], 0.0f);
}

// ---------------------------------------------------------------- launch
static inline size_t alignup(size_t x) { return (x + 255) & ~(size_t)255; }

extern "C" void kernel_launch(void* const* d_in, const int* in_sizes, int n_in,
                              void* d_out, int out_size, void* d_ws,
                              size_t ws_size, hipStream_t stream) {
  (void)in_sizes; (void)n_in; (void)out_size; (void)ws_size;

  const float*     x       = (const float*)d_in[0];
  const long long* ei      = (const long long*)d_in[1];
  const float*     Wself0  = (const float*)d_in[2];
  const float*     Wneigh0 = (const float*)d_in[3];
  const float*     b0      = (const float*)d_in[4];
  const float*     g0      = (const float*)d_in[5];
  const float*     be0     = (const float*)d_in[6];
  const float*     Wself1  = (const float*)d_in[7];
  const float*     Wneigh1 = (const float*)d_in[8];
  const float*     b1      = (const float*)d_in[9];
  const float*     g1      = (const float*)d_in[10];
  const float*     be1     = (const float*)d_in[11];
  float*           out     = (float*)d_out;

  char* ws = (char*)d_ws;
  const size_t SZ_XB  = (size_t)N_NODES * IN_DIM  * sizeof(__bf16);  // 25.6 MB
  const size_t SZ_H64 = (size_t)N_NODES * HID_DIM * sizeof(float);   // 25.6 MB
  const size_t SZ_H32 = (size_t)N_NODES * OUT_DIM * sizeof(float);   // 12.8 MB
  const size_t SZ_CNT = (size_t)N_NODES * sizeof(float);

  size_t o_xb   = 0;
  size_t o_S0   = alignup(o_xb + SZ_XB);
  size_t o_P0   = alignup(o_S0 + SZ_H64);
  size_t o_agg0 = alignup(o_P0 + SZ_H64);
  size_t o_cnt  = alignup(o_agg0 + SZ_H64);
  // aliases (lifetimes are disjoint under stream ordering):
  size_t o_h    = o_xb;              // bf16 h reuses xb (dead after gemm0)
  size_t o_S1   = o_S0;              // S1+P1 fit exactly in the S0 region
  size_t o_P1   = o_S0 + SZ_H32;
  size_t o_agg1 = o_P0;              // P0 dead after scatter0

  __bf16* xb   = (__bf16*)(ws + o_xb);
  float*  S0   = (float*)(ws + o_S0);
  float*  P0   = (float*)(ws + o_P0);
  float*  agg0 = (float*)(ws + o_agg0);
  float*  cnt  = (float*)(ws + o_cnt);
  __bf16* h    = (__bf16*)(ws + o_h);
  float*  S1   = (float*)(ws + o_S1);
  float*  P1   = (float*)(ws + o_P1);
  float*  agg1 = (float*)(ws + o_agg1);

  const long long nX = (long long)N_NODES * IN_DIM;

  // 1) x -> bf16
  cvt_bf16_kernel<<<(int)((nX / 4 + 255) / 256), 256, 0, stream>>>(x, xb, nX);

  // 2) zero cnt / agg0 (graph-capturable async memsets)
  hipMemsetAsync(cnt, 0, SZ_CNT, stream);
  hipMemsetAsync(agg0, 0, SZ_H64, stream);

  // 3) degrees
  degree_kernel<<<(N_EDGES + 255) / 256, 256, 0, stream>>>(ei, cnt, N_EDGES);

  // 4) layer-0 GEMMs: S0 = x@Wself0, P0 = x@Wneigh0   (WMMA bf16)
  {
    int tiles = N_NODES / 16;            // 6250
    int grid = (tiles + 7) / 8;          // 8 waves/block
    sage_gemm_kernel<IN_DIM, HID_DIM><<<grid, 256, 0, stream>>>(
        xb, Wself0, Wneigh0, S0, P0, N_NODES);
  }

  // 5) scatter projected neighbors: agg0[dst] += P0[src]  (64 f/edge)
  {
    long long thr = (long long)N_EDGES * (HID_DIM / 4);
    scatter_add_kernel<HID_DIM><<<(int)((thr + 255) / 256), 256, 0, stream>>>(
        P0, ei, agg0, N_EDGES);
  }

  // 6) combine + LN + ReLU -> h (bf16)
  combine_ln64_kernel<<<(N_NODES + 7) / 8, 256, 0, stream>>>(
      S0, agg0, cnt, b0, g0, be0, h, N_NODES);

  // 7) layer-1 GEMMs: S1 = h@Wself1, P1 = h@Wneigh1
  {
    int tiles = N_NODES / 16;
    int grid = (tiles + 7) / 8;
    sage_gemm_kernel<HID_DIM, OUT_DIM><<<grid, 256, 0, stream>>>(
        h, Wself1, Wneigh1, S1, P1, N_NODES);
  }

  // 8) zero agg1, scatter layer-1 (32 f/edge)
  hipMemsetAsync(agg1, 0, SZ_H32, stream);
  {
    long long thr = (long long)N_EDGES * (OUT_DIM / 4);
    scatter_add_kernel<OUT_DIM><<<(int)((thr + 255) / 256), 256, 0, stream>>>(
        P1, ei, agg1, N_EDGES);
  }

  // 9) final combine + LN + ReLU -> d_out (f32)
  combine_ln32_kernel<<<(N_NODES + 7) / 8, 256, 0, stream>>>(
      S1, agg1, cnt, b1, g1, be1, out, N_NODES);
}
